// FusionTransformer_21311627723015
// MI455X (gfx1250) — compile-verified, matrix operand order checked by reading, not run on hardware
//
#include <hip/hip_runtime.h>
#include <cstddef>
#include <cstdint>

// Fully-fused transformer block for MI455X (gfx1250, wave32, WMMA + pipelined TDM).
// Key insight: softmax over a singleton axis == 1, so attention output == v.
// z = y + gelu(LN2(y) @ W1 + bm1) @ W2 + bm2, y = (LN1(x) @ Wv + bv) @ Wp + bp.
//
// Weights are pre-converted once into 48 uniform f16 "LDS chunk images"
// (transposed [n][k], stride 40 halves, 10240 B each) laid out in exact
// consumption order in d_ws. The fused kernel double-buffers them with
// tensor_load_to_lds issued by wave 0, prefetching chunk i+1 while all
// 8 waves run WMMA on chunk i (one barrier per step).

#define BB   8
#define LL   8192
#define CC   384
#define DIN  128
#define HID  512
#define MT   32       // tokens per workgroup
#define NTHREADS 256  // 8 waves (wave32)
#define EPSF 1e-5f

// padded LDS strides (halves / floats), all row starts 16B aligned
#define SH_H_S 392    // 784 B = 49*16
#define WB_S   40     // 80 B  = 5*16
#define SH_V_S 136    // 272 B = 17*16
#define SH_Y_S 132    // floats, 528 B = 33*16
#define SH_T_S 136
#define SH_U_S 520    // 1040 B = 65*16
#define SH_Z_S 132    // floats (aliases sh_h region)

// chunk sequence: 0..11 Wv(kt) | 12..15 Wp(kt) | 16..31 W1(kt*4+ng) | 32..47 W2(kt)
#define CH128   (128 * WB_S)     // 5120 halves = 10240 B per chunk
#define NCHUNKS 48
#define SEQ_WV  0
#define SEQ_WP  12
#define SEQ_W1  16
#define SEQ_W2  32
#define WS_HALVES (NCHUNKS * CH128)   // 245760 halves = 491520 B

typedef __attribute__((ext_vector_type(16))) _Float16 v16h;
typedef __attribute__((ext_vector_type(8)))  float    v8f;
typedef __attribute__((ext_vector_type(4)))  unsigned v4u;
typedef __attribute__((ext_vector_type(8)))  int      v8i;
typedef __attribute__((ext_vector_type(4)))  int      v4i;

#if defined(__gfx1250__) && defined(__has_builtin)
# if __has_builtin(__builtin_amdgcn_tensor_load_to_lds)
#  define USE_TDM 1
# endif
#endif
#ifndef USE_TDM
# define USE_TDM 0
#endif
#if __has_include(<hip/amd_detail/amd_gfx1250_TDM.h>)
# define TDM_ARGS6 1
#else
# define TDM_ARGS6 0
#endif

__device__ __forceinline__ float wave_sum(float v) {
#pragma unroll
  for (int off = 16; off > 0; off >>= 1) v += __shfl_xor(v, off, 32);
  return v;
}

// Load one 16x32 f16 operand tile (A layout; B via transposed [n][k] storage).
__device__ __forceinline__ v16h load_tile16x32(const _Float16* base, int stride, int k0, int lane) {
  const int r  = lane & 15;
  const int hi = lane >> 4;
  const _Float16* p = base + r * stride + k0 + hi * 8;
  union { float4 f4[2]; v16h v; } u;
  u.f4[0] = *(const float4*)(p);
  u.f4[1] = *(const float4*)(p + 16);
  return u.v;
}

__device__ __forceinline__ v8f wmma_f16(v16h a, v16h b, v8f c) {
  return __builtin_amdgcn_wmma_f32_16x16x32_f16(false, a, false, b, (short)0, c, false, false);
}

#if USE_TDM
// Issue one fixed-size (CH128 halves) 1-D TDM copy ws -> LDS. Wave-0 only,
// caller gates by control flow (TDM ignores EXEC).
__device__ __forceinline__ void tdm_issue_chunk(const _Float16* gsrc, _Float16* ldst) {
  unsigned lds_addr = (unsigned)(unsigned long long)(uintptr_t)ldst; // low 32b = LDS offset
  unsigned long long ga = (unsigned long long)(uintptr_t)gsrc;
  constexpr unsigned dim0 = CH128;  // elements, data_size=2B
  v4u g0;                                     // count=1 | lds | global[56:0] | type=2
  g0[0] = 1u;
  g0[1] = lds_addr;
  g0[2] = (unsigned)ga;
  g0[3] = (unsigned)((ga >> 32) & 0x01FFFFFFu) | (2u << 30);
  v8i g1;                                     // data_size=1(2B), dims, 1-D tile
  g1[0] = (int)(1u << 16);                    // data_size
  g1[1] = (int)(dim0 << 16);                  // tensor_dim0 lo16
  g1[2] = (int)(1u << 16);                    // dim0 hi16=0 | tensor_dim1=1
  g1[3] = (int)(dim0 << 16);                  // dim1 hi16=0 | tile_dim0
  g1[4] = 0;                                  // tile_dim1=0, tile_dim2=0
  g1[5] = (int)dim0;                          // tensor_dim0_stride lo32
  g1[6] = 0;
  g1[7] = 0;
  v4i z4 = {0, 0, 0, 0};
#if TDM_ARGS6
  v8i z8 = {0, 0, 0, 0, 0, 0, 0, 0};
  __builtin_amdgcn_tensor_load_to_lds(g0, g1, z4, z4, z8, 0);
#else
  __builtin_amdgcn_tensor_load_to_lds(g0, g1, z4, z4, 0);
#endif
}
#endif

// Begin compute step `cidx`: make chunk cidx visible in wbuf[cidx&1] and
// prefetch chunk cidx+1 into the other buffer.
__device__ __forceinline__ void step_begin(int cidx, const _Float16* __restrict__ wsrc,
                                           _Float16* wbuf, int tid) {
#if USE_TDM
  if (tid < 32) __builtin_amdgcn_s_wait_tensorcnt(0);  // chunk cidx complete
  __syncthreads();                                     // publish; prev step fully done
  if (cidx + 1 < NCHUNKS && tid < 32)
    tdm_issue_chunk(wsrc + (size_t)(cidx + 1) * CH128, wbuf + ((cidx + 1) & 1) * CH128);
#else
  __syncthreads();
  {
    _Float16* dst = wbuf + (cidx & 1) * CH128;
    const _Float16* src = wsrc + (size_t)cidx * CH128;
    for (int off = tid * 8; off < CH128; off += NTHREADS * 8)
      *(float4*)(dst + off) = *(const float4*)(src + off);
    __syncthreads();
  }
#endif
}

// ---------------- prep: f32 weights -> f16 transposed chunk images ----------------
__global__ __launch_bounds__(128)
void prep_weights(const float* __restrict__ Wkv, const float* __restrict__ Wp,
                  const float* __restrict__ W1,  const float* __restrict__ W2,
                  _Float16* __restrict__ ws) {
  const int c = blockIdx.x;     // chunk id 0..47
  const int n = threadIdx.x;    // row-in-chunk 0..127
  const float* src;
  int kt, ldw, col;
  if (c < SEQ_WP)      { kt = c;              ldw = 2 * DIN; col = DIN + n;                src = Wkv; }
  else if (c < SEQ_W1) { kt = c - SEQ_WP;     ldw = DIN;     col = n;                      src = Wp;  }
  else if (c < SEQ_W2) { kt = (c - SEQ_W1) >> 2; ldw = HID;  col = ((c - SEQ_W1) & 3) * 128 + n; src = W1; }
  else                 { kt = c - SEQ_W2;     ldw = DIN;     col = n;                      src = W2;  }
  __attribute__((aligned(16))) _Float16 tmp[WB_S];
#pragma unroll
  for (int kk = 0; kk < 32; ++kk) tmp[kk] = (_Float16)src[(size_t)(kt * 32 + kk) * ldw + col];
#pragma unroll
  for (int kk = 32; kk < WB_S; ++kk) tmp[kk] = (_Float16)0.f;
  _Float16* dst = ws + (size_t)c * CH128 + n * WB_S;
#pragma unroll
  for (int j = 0; j < WB_S / 8; ++j)
    *(float4*)(dst + j * 8) = *(const float4*)(tmp + j * 8);
}

// ---------------- fused transformer block ----------------
__global__ __launch_bounds__(NTHREADS)
void fused_block_kernel(const float* __restrict__ x,
                        const float* __restrict__ g1, const float* __restrict__ b1,
                        const float* __restrict__ bkv,
                        const float* __restrict__ bp,
                        const float* __restrict__ g2,  const float* __restrict__ b2,
                        const float* __restrict__ bm1, const float* __restrict__ bm2,
                        const _Float16* __restrict__ wsrc,
                        float* __restrict__ out) {
  __shared__ __align__(16) unsigned char sh_hz[MT * SH_H_S * 2]; // sh_h (f16) then sh_z (f32)
  __shared__ _Float16 wbuf[2 * CH128];
  __shared__ _Float16 sh_v[MT * SH_V_S];
  __shared__ float    sh_y[MT * SH_Y_S];
  __shared__ _Float16 sh_t[MT * SH_T_S];
  __shared__ _Float16 sh_u[MT * SH_U_S];
  _Float16* sh_h = (_Float16*)sh_hz;
  float*    sh_z = (float*)sh_hz;   // reused after stage 1

  const int tid    = threadIdx.x;
  const int lane   = tid & 31;
  const int wave   = tid >> 5;
  const int token0 = blockIdx.x * MT;
  const int n0     = lane & 15;
  const int hi     = lane >> 4;

  union H4 { _Float16 h[4]; float2 f2; };

#if USE_TDM
  if (tid < 32) tdm_issue_chunk(wsrc, wbuf);   // prefetch chunk 0 behind LN1
#endif

  // ---- Stage 0: LN1(x) -> sh_h (f16). Lane owns 12 contiguous columns. ----
  {
    const int c0 = lane * 12;
    float4 ga_[3], ba_[3];
#pragma unroll
    for (int j = 0; j < 3; ++j) {
      ga_[j] = *(const float4*)(g1 + c0 + 4 * j);
      ba_[j] = *(const float4*)(b1 + c0 + 4 * j);
    }
    for (int r = wave; r < MT; r += 8) {
      const float* xr = x + (size_t)(token0 + r) * CC;
      float4 xv[3];
      float s = 0.f, s2 = 0.f;
#pragma unroll
      for (int j = 0; j < 3; ++j) {
        xv[j] = *(const float4*)(xr + c0 + 4 * j);
#pragma unroll
        for (int e = 0; e < 4; ++e) { float v = ((const float*)&xv[j])[e]; s += v; s2 += v * v; }
      }
      s  = wave_sum(s);
      s2 = wave_sum(s2);
      float mean = s * (1.f / CC);
      float rs   = rsqrtf(s2 * (1.f / CC) - mean * mean + EPSF);
#pragma unroll
      for (int j = 0; j < 3; ++j) {
        H4 u;
#pragma unroll
        for (int e = 0; e < 4; ++e) {
          float v = ((const float*)&xv[j])[e];
          u.h[e] = (_Float16)((v - mean) * rs * ((const float*)&ga_[j])[e] + ((const float*)&ba_[j])[e]);
        }
        *(float2*)(&sh_h[r * SH_H_S + c0 + 4 * j]) = u.f2;
      }
    }
  }

  // ---- Stage 1: V = H(32x384) @ Wv + bv ----
  {
    v8f acc0 = {}, acc1 = {};
    for (int kt = 0; kt < CC / 32; ++kt) {
      step_begin(SEQ_WV + kt, wsrc, wbuf, tid);
      _Float16* wb = wbuf + ((SEQ_WV + kt) & 1) * CH128;
      v16h a0 = load_tile16x32(sh_h,               SH_H_S, kt * 32, lane);
      v16h a1 = load_tile16x32(sh_h + 16 * SH_H_S, SH_H_S, kt * 32, lane);
      v16h bw = load_tile16x32(wb + wave * 16 * WB_S, WB_S, 0, lane);
      acc0 = wmma_f16(a0, bw, acc0);
      acc1 = wmma_f16(a1, bw, acc1);
    }
    int col = wave * 16 + n0;
    float bias = bkv[DIN + col];
#pragma unroll
    for (int r = 0; r < 8; ++r) {
      sh_v[(hi * 8 + r) * SH_V_S + col]      = (_Float16)(acc0[r] + bias);
      sh_v[(16 + hi * 8 + r) * SH_V_S + col] = (_Float16)(acc1[r] + bias);
    }
  }

  // ---- Stage 2: Y = V(32x128) @ Wp + bp (f32 in LDS) ----
  {
    v8f acc0 = {}, acc1 = {};
    for (int kt = 0; kt < DIN / 32; ++kt) {
      step_begin(SEQ_WP + kt, wsrc, wbuf, tid);
      _Float16* wb = wbuf + ((SEQ_WP + kt) & 1) * CH128;
      v16h a0 = load_tile16x32(sh_v,               SH_V_S, kt * 32, lane);
      v16h a1 = load_tile16x32(sh_v + 16 * SH_V_S, SH_V_S, kt * 32, lane);
      v16h bw = load_tile16x32(wb + wave * 16 * WB_S, WB_S, 0, lane);
      acc0 = wmma_f16(a0, bw, acc0);
      acc1 = wmma_f16(a1, bw, acc1);
    }
    int col = wave * 16 + n0;
    float bias = bp[col];
#pragma unroll
    for (int r = 0; r < 8; ++r) {
      sh_y[(hi * 8 + r) * SH_Y_S + col]      = acc0[r] + bias;
      sh_y[(16 + hi * 8 + r) * SH_Y_S + col] = acc1[r] + bias;
    }
  }
  __syncthreads();   // publish sh_y before LN2

  // ---- Stage 3: T = LN2(Y) -> sh_t (f16). Lane owns 4 contiguous columns. ----
  {
    const int c0 = lane * 4;
    float4 gv = *(const float4*)(g2 + c0);
    float4 bv = *(const float4*)(b2 + c0);
    for (int r = wave; r < MT; r += 8) {
      float4 yv = *(const float4*)(&sh_y[r * SH_Y_S + c0]);
      float s = 0.f, s2 = 0.f;
#pragma unroll
      for (int e = 0; e < 4; ++e) { float v = ((const float*)&yv)[e]; s += v; s2 += v * v; }
      s  = wave_sum(s);
      s2 = wave_sum(s2);
      float mean = s * (1.f / DIN);
      float rs   = rsqrtf(s2 * (1.f / DIN) - mean * mean + EPSF);
      H4 u;
#pragma unroll
      for (int e = 0; e < 4; ++e) {
        float v = ((const float*)&yv)[e];
        u.h[e] = (_Float16)((v - mean) * rs * ((const float*)&gv)[e] + ((const float*)&bv)[e]);
      }
      *(float2*)(&sh_t[r * SH_T_S + c0]) = u.f2;
    }
  }

  // ---- Stage 4: U = gelu(T(32x128) @ W1 + bm1); chunks cover 128-col groups ----
  {
    v8f acc[2][4] = {};
    v16h a0, a1;
    for (int kt = 0; kt < DIN / 32; ++kt) {
#pragma unroll
      for (int ng = 0; ng < 4; ++ng) {
        const int cidx = SEQ_W1 + kt * 4 + ng;
        step_begin(cidx, wsrc, wbuf, tid);
        if (ng == 0) {
          a0 = load_tile16x32(sh_t,               SH_T_S, kt * 32, lane);
          a1 = load_tile16x32(sh_t + 16 * SH_T_S, SH_T_S, kt * 32, lane);
        }
        _Float16* wb = wbuf + (cidx & 1) * CH128;
        v16h bw = load_tile16x32(wb + wave * 16 * WB_S, WB_S, 0, lane);
        acc[0][ng] = wmma_f16(a0, bw, acc[0][ng]);
        acc[1][ng] = wmma_f16(a1, bw, acc[1][ng]);
      }
    }
#pragma unroll
    for (int ng = 0; ng < 4; ++ng) {
      int col = ng * 128 + wave * 16 + n0;
      float bias = bm1[col];
#pragma unroll
      for (int m = 0; m < 2; ++m) {
#pragma unroll
        for (int r = 0; r < 8; ++r) {
          float v = acc[m][ng][r] + bias;
          float g = 0.5f * v * (1.f + erff(v * 0.70710678118654752f)); // exact gelu
          sh_u[(m * 16 + hi * 8 + r) * SH_U_S + col] = (_Float16)g;
        }
      }
    }
  }

  // ---- Stage 5: Z = Y + U(32x512) @ W2 + bm2 ----
  {
    v8f acc0 = {}, acc1 = {};
    for (int kt = 0; kt < HID / 32; ++kt) {
      step_begin(SEQ_W2 + kt, wsrc, wbuf, tid);
      _Float16* wb = wbuf + ((SEQ_W2 + kt) & 1) * CH128;
      v16h a0 = load_tile16x32(sh_u,               SH_U_S, kt * 32, lane);
      v16h a1 = load_tile16x32(sh_u + 16 * SH_U_S, SH_U_S, kt * 32, lane);
      v16h bw = load_tile16x32(wb + wave * 16 * WB_S, WB_S, 0, lane);
      acc0 = wmma_f16(a0, bw, acc0);
      acc1 = wmma_f16(a1, bw, acc1);
    }
    int col = wave * 16 + n0;
    float bias = bm2[col];
#pragma unroll
    for (int r = 0; r < 8; ++r) {
      int row0 = hi * 8 + r;
      int row1 = 16 + hi * 8 + r;
      sh_z[row0 * SH_Z_S + col] = sh_y[row0 * SH_Y_S + col] + acc0[r] + bias;
      sh_z[row1 * SH_Z_S + col] = sh_y[row1 * SH_Y_S + col] + acc1[r] + bias;
    }
  }
  __syncthreads();

  // ---- Coalesced writeback: 8 threads per row, float4 stores ----
  {
    const int row = tid >> 3;
    const int seg = tid & 7;
    const float* zr = sh_z + row * SH_Z_S + seg * 16;
    float* orow = out + (size_t)(token0 + row) * DIN + seg * 16;
#pragma unroll
    for (int j = 0; j < 4; ++j)
      *(float4*)(orow + j * 4) = *(const float4*)(zr + j * 4);
  }
}

extern "C" void kernel_launch(void* const* d_in, const int* in_sizes, int n_in,
                              void* d_out, int out_size, void* d_ws, size_t ws_size,
                              hipStream_t stream) {
  // 0 x, 1 g1, 2 b1, 3 Wq(unused), 4 bq(unused), 5 Wkv, 6 bkv, 7 rpb(unused),
  // 8 Wp, 9 bp, 10 g2, 11 b2, 12 W1, 13 bm1, 14 W2, 15 bm2
  const float* x   = (const float*)d_in[0];
  const float* g1  = (const float*)d_in[1];
  const float* b1  = (const float*)d_in[2];
  const float* Wkv = (const float*)d_in[5];
  const float* bkv = (const float*)d_in[6];
  const float* Wp  = (const float*)d_in[8];
  const float* bp  = (const float*)d_in[9];
  const float* g2  = (const float*)d_in[10];
  const float* b2  = (const float*)d_in[11];
  const float* W1  = (const float*)d_in[12];
  const float* bm1 = (const float*)d_in[13];
  const float* W2  = (const float*)d_in[14];
  const float* bm2 = (const float*)d_in[15];
  float* out = (float*)d_out;
  _Float16* ws = (_Float16*)d_ws;

  hipLaunchKernelGGL(prep_weights, dim3(NCHUNKS), dim3(128), 0, stream, Wkv, Wp, W1, W2, ws);

  const int tokens = BB * LL;
  hipLaunchKernelGGL(fused_block_kernel, dim3(tokens / MT), dim3(NTHREADS), 0, stream,
                     x, g1, b1, bkv, bp, g2, b2, bm1, bm2, (const _Float16*)ws, out);
  (void)in_sizes; (void)n_in; (void)out_size; (void)ws_size;
}
